// KMeansBatchNorm_38594576122529
// MI455X (gfx1250) — matrix operand order repaired and checked by reading, not compiled, not run on hardware
//
#include <hip/hip_runtime.h>

typedef __attribute__((ext_vector_type(16))) _Float16 v16h;
typedef __attribute__((ext_vector_type(8)))  float    v8f;

#define B_TOT 512
#define C_CH  64
#define HW    1024
#define D_TOT 65536   // C_CH * HW
#define K_CL  8
#define EPSV  1e-5f

// ---------------------------------------------------------------------------
// Kernel A: cnorm[k] = |c_k|^2
// ---------------------------------------------------------------------------
__global__ void kmbn_cnorm(const float* __restrict__ cmat, float* __restrict__ cnorm) {
  __shared__ float red[256];
  const int k = blockIdx.x;
  const float4* r = (const float4*)(cmat + (size_t)k * D_TOT);
  float s = 0.f;
  for (int i = threadIdx.x; i < D_TOT / 4; i += 256) {
    float4 v = r[i];
    s += v.x * v.x + v.y * v.y + v.z * v.z + v.w * v.w;
  }
  red[threadIdx.x] = s;
  __syncthreads();
  for (int off = 128; off > 0; off >>= 1) {
    if (threadIdx.x < off) red[threadIdx.x] += red[threadIdx.x + off];
    __syncthreads();
  }
  if (threadIdx.x == 0) cnorm[k] = red[0];
}

// ---------------------------------------------------------------------------
// Kernel B: fused WMMA dot products (x . c_k) + per-(sample,channel) sum/sumsq.
// Grid: (32 sample-groups, 8 D-splits). Block: 256 threads = 8 wave32.
// Each block: samples b0..b0+15, channels sp*8..sp*8+7 (k-range sp*8192..+8192).
// Wave w handles k-chunks kk = sp*8192 + (cl*4+j)*256 + w*32 (32-wide WMMA steps).
// ---------------------------------------------------------------------------
__global__ void __launch_bounds__(256)
kmbn_dots_sums(const float* __restrict__ x, const float* __restrict__ cmat,
               float* __restrict__ dots, float* __restrict__ sb1,
               float* __restrict__ sb2) {
  __shared__ float s1L[8][8][16];   // [wave][local ch][row]
  __shared__ float s2L[8][8][16];
  __shared__ float dotL[8 * 256];   // per-wave accumulator dump

  const int g    = blockIdx.x;            // sample group
  const int sp   = blockIdx.y;            // D split
  const int w    = threadIdx.x >> 5;
  const int lane = threadIdx.x & 31;
  const int row  = lane & 15;
  const int hi   = lane >> 4;
  const int b0   = g * 16;

  const float* xrow = x + (size_t)(b0 + row) * D_TOT;
  const float* crow = cmat + (size_t)(row & 7) * D_TOT;

  v8f acc = {};

  for (int cl = 0; cl < 8; ++cl) {        // local channel within this split
    float s1v = 0.f, s2v = 0.f;
#pragma unroll
    for (int j = 0; j < 4; ++j) {
      const int kk = sp * 8192 + (cl * 4 + j) * 256 + w * 32;

      // A fragment: 16x32 f16 layout -> lane holds K = {kk+hi*8+0..7, kk+16+hi*8+0..7}
      const int oa0 = kk + hi * 8;
      const int oa1 = kk + 16 + hi * 8;
      float4 a0 = *(const float4*)(xrow + oa0);
      float4 a1 = *(const float4*)(xrow + oa0 + 4);
      float4 a2 = *(const float4*)(xrow + oa1);
      float4 a3 = *(const float4*)(xrow + oa1 + 4);

      s1v += a0.x + a0.y + a0.z + a0.w + a1.x + a1.y + a1.z + a1.w
           + a2.x + a2.y + a2.z + a2.w + a3.x + a3.y + a3.z + a3.w;
      s2v += a0.x*a0.x + a0.y*a0.y + a0.z*a0.z + a0.w*a0.w
           + a1.x*a1.x + a1.y*a1.y + a1.z*a1.z + a1.w*a1.w
           + a2.x*a2.x + a2.y*a2.y + a2.z*a2.z + a2.w*a2.w
           + a3.x*a3.x + a3.y*a3.y + a3.z*a3.z + a3.w*a3.w;

      v16h af;
      af[0] = (_Float16)a0.x; af[1] = (_Float16)a0.y; af[2]  = (_Float16)a0.z; af[3]  = (_Float16)a0.w;
      af[4] = (_Float16)a1.x; af[5] = (_Float16)a1.y; af[6]  = (_Float16)a1.z; af[7]  = (_Float16)a1.w;
      af[8] = (_Float16)a2.x; af[9] = (_Float16)a2.y; af[10] = (_Float16)a2.z; af[11] = (_Float16)a2.w;
      af[12] = (_Float16)a3.x; af[13] = (_Float16)a3.y; af[14] = (_Float16)a3.z; af[15] = (_Float16)a3.w;

      // B fragment: 32x16 f16 layout -> lane holds col N=row, K = {kk+hi*8+0..7, kk+16+hi*8+0..7}.
      // Columns 8..15 (no centroid) are zero.
      v16h bf;
      if (row < 8) {
        const int ob0 = kk + hi * 8;
        const int ob1 = kk + 16 + hi * 8;
        float4 c0 = *(const float4*)(crow + ob0);
        float4 c1 = *(const float4*)(crow + ob0 + 4);
        float4 c2 = *(const float4*)(crow + ob1);
        float4 c3 = *(const float4*)(crow + ob1 + 4);
        bf[0] = (_Float16)c0.x; bf[1] = (_Float16)c0.y; bf[2]  = (_Float16)c0.z; bf[3]  = (_Float16)c0.w;
        bf[4] = (_Float16)c1.x; bf[5] = (_Float16)c1.y; bf[6]  = (_Float16)c1.z; bf[7]  = (_Float16)c1.w;
        bf[8] = (_Float16)c2.x; bf[9] = (_Float16)c2.y; bf[10] = (_Float16)c2.z; bf[11] = (_Float16)c2.w;
        bf[12] = (_Float16)c3.x; bf[13] = (_Float16)c3.y; bf[14] = (_Float16)c3.z; bf[15] = (_Float16)c3.w;
      } else {
#pragma unroll
        for (int e = 0; e < 16; ++e) bf[e] = (_Float16)0.f;
      }

      acc = __builtin_amdgcn_wmma_f32_16x16x32_f16(
          /*neg_a=*/false, af, /*neg_b=*/false, bf,
          /*c_mod=*/(short)0, acc, /*reuse_a=*/false, /*reuse_b=*/false);
    }
    // fold the two lane halves of each row, stash per-(wave,channel,row) sums
    s1v += __shfl_xor(s1v, 16, 32);
    s2v += __shfl_xor(s2v, 16, 32);
    if (hi == 0) { s1L[w][cl][row] = s1v; s2L[w][cl][row] = s2v; }
  }

#pragma unroll
  for (int v = 0; v < 8; ++v) dotL[w * 256 + v * 32 + lane] = acc[v];
  __syncthreads();

  const int t = threadIdx.x;
  // cross-wave reduce of channel sums -> sb1/sb2 (exclusive per D-split, no atomics)
  if (t < 128) {
    const int ch = t >> 4, r2 = t & 15;
    float v1 = 0.f, v2 = 0.f;
#pragma unroll
    for (int ww = 0; ww < 8; ++ww) { v1 += s1L[ww][ch][r2]; v2 += s2L[ww][ch][r2]; }
    const size_t idx = (size_t)(b0 + r2) * C_CH + sp * 8 + ch;
    sb1[idx] = v1;
    sb2[idx] = v2;
  }
  // cross-wave reduce of the 16x16 accumulator; C/D layout: N=lane&15, M=vgpr+8*(lane>=16)
  {
    float dsum = 0.f;
#pragma unroll
    for (int ww = 0; ww < 8; ++ww) dsum += dotL[ww * 256 + t];
    const int l2 = t & 31, vg = t >> 5;
    const int N = l2 & 15;
    const int M = vg + ((l2 >> 4) << 3);
    if (N < K_CL) atomicAdd(&dots[(size_t)(b0 + M) * K_CL + N], dsum);
  }
}

// ---------------------------------------------------------------------------
// Kernel C: argmin_k ( |c_k|^2 - 2 * dot[b,k] )
// ---------------------------------------------------------------------------
__global__ void kmbn_assign(const float* __restrict__ dots,
                            const float* __restrict__ cnorm,
                            int* __restrict__ assign_out) {
  const int b = blockIdx.x * blockDim.x + threadIdx.x;
  if (b >= B_TOT) return;
  float best = cnorm[0] - 2.f * dots[b * K_CL + 0];
  int bi = 0;
#pragma unroll
  for (int k = 1; k < K_CL; ++k) {
    const float sc = cnorm[k] - 2.f * dots[b * K_CL + k];
    if (sc < best) { best = sc; bi = k; }
  }
  assign_out[b] = bi;
}

// ---------------------------------------------------------------------------
// Kernel D: per-(cluster,channel) BN stats -> fused scale/bias
// ---------------------------------------------------------------------------
__global__ void kmbn_stats(const int* __restrict__ assign_in,
                           const float* __restrict__ sb1, const float* __restrict__ sb2,
                           const float* __restrict__ gamma, const float* __restrict__ beta,
                           float* __restrict__ scale, float* __restrict__ bias) {
  const int k = blockIdx.x;
  const int ch = threadIdx.x;        // 64 threads
  float s1 = 0.f, s2 = 0.f;
  int cnt = 0;
  for (int b = 0; b < B_TOT; ++b) {
    if (assign_in[b] == k) {
      s1 += sb1[b * C_CH + ch];
      s2 += sb2[b * C_CH + ch];
      ++cnt;
    }
  }
  const float n = fmaxf((float)cnt * (float)HW, 1.0f);
  const float mean = s1 / n;
  const float var  = s2 / n - mean * mean;
  const float inv  = rsqrtf(var + EPSV);
  const float g  = gamma[k * C_CH + ch];
  const float be = beta[k * C_CH + ch];
  const float sc = inv * g;
  scale[k * C_CH + ch] = sc;
  bias[k * C_CH + ch]  = be - mean * sc;
}

// ---------------------------------------------------------------------------
// Kernel E: out = x * scale[assign[b],c] + bias[assign[b],c]
// One block per (b, c) plane of 1024 floats; 256 threads x 1 float4.
// ---------------------------------------------------------------------------
__global__ void __launch_bounds__(256)
kmbn_normalize(const float* __restrict__ x, const int* __restrict__ assign_in,
               const float* __restrict__ scale, const float* __restrict__ bias,
               float* __restrict__ out) {
  const int plane = blockIdx.x;            // b*64 + c
  const int b  = plane >> 6;
  const int ch = plane & 63;
  const int a  = assign_in[b];
  const float sc = scale[a * C_CH + ch];
  const float bi = bias[a * C_CH + ch];
  const float4* xp = (const float4*)(x + (size_t)plane * HW);
  float4*       op = (float4*)(out + (size_t)plane * HW);
  float4 v = xp[threadIdx.x];
  float4 r;
  r.x = v.x * sc + bi;
  r.y = v.y * sc + bi;
  r.z = v.z * sc + bi;
  r.w = v.w * sc + bi;
  op[threadIdx.x] = r;
}

// ---------------------------------------------------------------------------
extern "C" void kernel_launch(void* const* d_in, const int* in_sizes, int n_in,
                              void* d_out, int out_size, void* d_ws, size_t ws_size,
                              hipStream_t stream) {
  const float* x     = (const float*)d_in[0];   // [512,64,32,32]
  const float* cmat  = (const float*)d_in[1];   // [8, 65536]
  const float* gamma = (const float*)d_in[2];   // [8, 64]
  const float* beta  = (const float*)d_in[3];   // [8, 64]
  float* out = (float*)d_out;

  // workspace layout (floats / ints, all 4-byte aligned)
  float* cnorm     = (float*)d_ws;                    // 8
  float* dots      = cnorm + 8;                       // 512*8
  int*   assign_ws = (int*)(dots + B_TOT * K_CL);     // 512
  float* sb1       = (float*)(assign_ws + B_TOT);     // 512*64
  float* sb2       = sb1 + B_TOT * C_CH;              // 512*64
  float* scale     = sb2 + B_TOT * C_CH;              // 8*64
  float* bias      = scale + K_CL * C_CH;             // 8*64

  // dots is atomically accumulated -> must be zeroed every call (graph-safe)
  hipMemsetAsync(dots, 0, (size_t)B_TOT * K_CL * sizeof(float), stream);

  kmbn_cnorm<<<K_CL, 256, 0, stream>>>(cmat, cnorm);
  kmbn_dots_sums<<<dim3(B_TOT / 16, 8), 256, 0, stream>>>(x, cmat, dots, sb1, sb2);
  kmbn_assign<<<2, 256, 0, stream>>>(dots, cnorm, assign_ws);
  kmbn_stats<<<K_CL, C_CH, 0, stream>>>(assign_ws, sb1, sb2, gamma, beta, scale, bias);
  kmbn_normalize<<<B_TOT * C_CH, 256, 0, stream>>>(x, assign_ws, scale, bias, out);
}
